// BlockSparseSelfAttention_43181601194862
// MI455X (gfx1250) — compile-verified, matrix operand order checked by reading, not run on hardware
//
#include <hip/hip_runtime.h>
#include <hip/hip_bf16.h>

// ---------------------------------------------------------------------------
// BlockSparseSelfAttention for MI455X (gfx1250, wave32, WMMA 16x16x32 f16).
// B=2, N=2048, UNITS=1024, H=16, D=64, NB=64.
// f16 storage / f32 WMMA accumulation; softmax scale folded into Q;
// flash-attention streaming with online softmax; additive -1e9 mask penalties.
// Weights are pre-transposed (N-major) so every WMMA operand everywhere is a
// row-contiguous 2x b128 fragment load. GEMM B-tiles are staged into LDS by
// the Tensor Data Mover (tensor_load_to_lds + s_wait_tensorcnt), using TDM's
// LDS pad feature to produce a bank-conflict-free 40-half row stride.
// ---------------------------------------------------------------------------

typedef __attribute__((ext_vector_type(16))) _Float16 v16h;
typedef __attribute__((ext_vector_type(8)))  float    v8f;
typedef __attribute__((ext_vector_type(4)))  unsigned int v4u;
typedef __attribute__((ext_vector_type(8)))  int      v8i_;
typedef __attribute__((ext_vector_type(4)))  int      v4i_;

#define BATCH     2
#define SEQ       2048
#define UNITS_    1024
#define NHEADS    16
#define HDIM      64
#define NBLK      64
#define MROWS     (BATCH * SEQ)       // 4096
#define NEG_INF_F (-1.0e9f)
#define LDST      40                  // padded LDS row stride (halves)

// Row-major 16x32 fragment: per lane, two contiguous 8-half (16B) runs at
// k = 8*half and k = 16 + 8*half (ISA 7.12.2 16-bit A-operand layout).
__device__ __forceinline__ v16h load_frag_rowmajor(const _Float16* src, int ld) {
  const int lane = threadIdx.x & 31;
  const int m = lane & 15, half = lane >> 4;
  const _Float16* p = src + (size_t)m * ld + 8 * half;
  v16h a;
  *reinterpret_cast<uint4*>(&a)       = *reinterpret_cast<const uint4*>(p);
  *(reinterpret_cast<uint4*>(&a) + 1) = *reinterpret_cast<const uint4*>(p + 16);
  return a;
}

// The TDM writes LDS behind the compiler's back. Escape the LDS pointer into
// opaque asm with a memory clobber so loads from it cannot be treated as
// reads of never-written (undef) memory, nor hoisted out of the K loop.
__device__ __forceinline__ void lds_written_fence(void* p) {
  asm volatile("" : "+v"(p) : : "memory");
}

// Issue one TDM load: 2-D tile (tile_dim0=32 halves contiguous, tile_dim1=64
// rows, row stride = ld halves) from global f16 -> LDS offset 0, with LDS
// padding of 4 DWORDs every 16 DWORDs (row stride 32 -> 40 halves).
__device__ __forceinline__ void tdm_load_wtile(const _Float16* gsrc, int ld) {
  const unsigned long long ga = (unsigned long long)(uintptr_t)gsrc;
  v4u g0 = { 1u,                                  // count=1, user descriptor
             0u,                                  // lds_addr = 0 (sole LDS obj)
             (unsigned int)ga,                    // global_addr[31:0]
             (unsigned int)((ga >> 32) & 0x01FFFFFFu) | (2u << 30) }; // type=2
  // flags: data_size=1(2B), pad_enable, pad_interval=3(16 DW), pad_amount=3(4 DW)
  v8i_ g1 = { (int)0x06D10000u,
              (int)(32u << 16),                   // tensor_dim0 = 32
              (int)(64u << 16),                   // tensor_dim1 = 64
              (int)(32u << 16),                   // tile_dim0   = 32
              64,                                 // tile_dim1   = 64
              ld,                                 // tensor_dim0_stride
              0, 0 };
  v4i_ gz = { 0, 0, 0, 0 };
#if __clang_major__ >= 23
  v8i_ gz8 = { 0, 0, 0, 0, 0, 0, 0, 0 };
  __builtin_amdgcn_tensor_load_to_lds(g0, g1, gz, gz, gz8, 0);
#else
  __builtin_amdgcn_tensor_load_to_lds(g0, g1, gz, gz, 0);
#endif
  __builtin_amdgcn_s_wait_tensorcnt(0);
}

// ---------------------------------------------------------------------------
// fp32 -> fp16 conversion
// ---------------------------------------------------------------------------
__global__ __launch_bounds__(256)
void f32_to_f16_kernel(const float* __restrict__ src, _Float16* __restrict__ dst, int n) {
  int i = blockIdx.x * blockDim.x + threadIdx.x;
  if (i < n) dst[i] = (_Float16)src[i];
}

// LDS-tiled transpose + convert: src f32 (rows x cols) -> dst f16 (cols x rows)
__global__ __launch_bounds__(256)
void transpose_f32_to_f16_kernel(const float* __restrict__ src, _Float16* __restrict__ dst,
                                 int rows, int cols) {
  __shared__ float tile[32][33];
  const int bx = blockIdx.x * 32;   // col base in src
  const int by = blockIdx.y * 32;   // row base in src
  const int tx = threadIdx.x & 31;
  const int ty = threadIdx.x >> 5;  // 0..7
#pragma unroll
  for (int i = 0; i < 32; i += 8)
    tile[ty + i][tx] = src[(size_t)(by + ty + i) * cols + bx + tx];
  __syncthreads();
#pragma unroll
  for (int i = 0; i < 32; i += 8)
    dst[(size_t)(bx + ty + i) * rows + by + tx] = (_Float16)tile[tx][ty + i];
}

// one-hot blockB (BN x 64) -> int block id
__global__ __launch_bounds__(256)
void block_id_kernel(const float* __restrict__ blockB, int* __restrict__ ids, int bn) {
  int i = blockIdx.x * blockDim.x + threadIdx.x;
  if (i >= bn) return;
  const float* row = blockB + (size_t)i * NBLK;
  int id = 0;
#pragma unroll 8
  for (int j = 0; j < NBLK; ++j)
    if (row[j] > 0.5f) id = j;
  ids[i] = id;
}

// ---------------------------------------------------------------------------
// GEMM: C[M,Nc] = A[M,K] * Wt^T, with Wt given TRANSPOSED (Nc x K row-major).
// 64x64 tile per 128-thread (4-wave) block. The 64x32 Wt tile is staged into
// LDS by the Tensor Data Mover (wave 0 issues, s_wait_tensorcnt, barrier).
// outMode: 0 = f32 row-major              (final @Wo)
//          2 = f16 head-major (B,H,N,D)   (Q, K)
//          3 = f16 transposed head-major (B,H,D,N)  (V)
// ---------------------------------------------------------------------------
__global__ __launch_bounds__(128)
void gemm_f16_kernel(const _Float16* __restrict__ A, const _Float16* __restrict__ Wt,
                     float* __restrict__ outF, _Float16* __restrict__ outH,
                     int M, int K, int Nc, int outMode, float scale) {
  __shared__ alignas(16) _Float16 ldsWT[64 * LDST];   // [n][k], TDM-padded
  const int wave = threadIdx.x >> 5;
  const int lane = threadIdx.x & 31;
  const int tileM = blockIdx.x * 64;
  const int tileN = blockIdx.y * 64;
  const int rowBase = tileM + wave * 16;

  v8f acc[4] = {{}, {}, {}, {}};

  for (int k0 = 0; k0 < K; k0 += 32) {
    if (wave == 0)
      tdm_load_wtile(Wt + (size_t)tileN * K + k0, K);
    lds_written_fence(&ldsWT[0]);    // TDM wrote LDS; loads must be real
    if (k0 + 32 < K)
      __builtin_prefetch(Wt + (size_t)tileN * K + k0 + 32, 0, 1);
    __syncthreads();

    v16h afrag = load_frag_rowmajor(A + (size_t)rowBase * K + k0, K);
#pragma unroll
    for (int j = 0; j < 4; ++j) {
      v16h bfrag = load_frag_rowmajor(&ldsWT[(j * 16) * LDST], LDST);
      acc[j] = __builtin_amdgcn_wmma_f32_16x16x32_f16(
          false, afrag, false, bfrag, (short)0, acc[j], false, false);
    }
    __syncthreads();
  }

  const int half = lane >> 4;
  const int nl = lane & 15;
#pragma unroll
  for (int j = 0; j < 4; ++j) {
#pragma unroll
    for (int r = 0; r < 8; ++r) {
      const int gm = rowBase + r + 8 * half;
      const int gn = tileN + j * 16 + nl;
      const float v = acc[j][r] * scale;
      if (outMode == 0) {
        outF[(size_t)gm * Nc + gn] = v;
      } else {
        const int b = gm / SEQ, n = gm % SEQ;
        const int h = gn / HDIM, d = gn % HDIM;
        if (outMode == 2)
          outH[(((size_t)b * NHEADS + h) * SEQ + n) * HDIM + d] = (_Float16)v;
        else // 3: V transposed (B,H,D,N)
          outH[(((size_t)b * NHEADS + h) * HDIM + d) * SEQ + n] = (_Float16)v;
      }
    }
  }
}

// ---------------------------------------------------------------------------
// Flash attention: one wave per (b, h, 16-row tile). Streams 32 keys/iter.
// Q carries the 1/sqrt(D) scale. V is pre-transposed (B,H,D,N) so the P@V
// B-fragment is two contiguous b128 loads along the key axis.
// ---------------------------------------------------------------------------
__global__ __launch_bounds__(32)
void flash_attn_kernel(const _Float16* __restrict__ Q, const _Float16* __restrict__ Km,
                       const _Float16* __restrict__ Vt, const int* __restrict__ blockId,
                       const float* __restrict__ nodal, _Float16* __restrict__ outAttn) {
  __shared__ alignas(16) _Float16 ldsP[16 * 32];
  const int lane = threadIdx.x & 31;
  const int nl = lane & 15, half = lane >> 4;

  const int mtiles = SEQ / 16;                       // 128
  const int bh = blockIdx.x / mtiles;
  const int b = bh / NHEADS;
  const int h = bh % NHEADS;
  const int rowBase = (blockIdx.x % mtiles) * 16;

  const _Float16* Qh  = Q  + (((size_t)b * NHEADS + h) * SEQ + rowBase) * HDIM;
  const _Float16* Kh  = Km + ((size_t)b * NHEADS + h) * SEQ * HDIM;
  const _Float16* Vth = Vt + ((size_t)b * NHEADS + h) * (size_t)HDIM * SEQ;

  v16h qf[2];
  qf[0] = load_frag_rowmajor(Qh, HDIM);
  qf[1] = load_frag_rowmajor(Qh + 32, HDIM);

  int rowblk[8];
#pragma unroll
  for (int r = 0; r < 8; ++r)
    rowblk[r] = blockId[b * SEQ + rowBase + r + 8 * half];

  float mrow[8], lrow[8];
#pragma unroll
  for (int r = 0; r < 8; ++r) { mrow[r] = -INFINITY; lrow[r] = 0.0f; }
  v8f acc[4] = {{}, {}, {}, {}};

  for (int kb = 0; kb < SEQ; kb += 32) {
    const int   cb   = blockId[b * SEQ + kb + lane];
    const float cnod = nodal[b * SEQ + kb + lane];

    // Scores: S[16x32] as two 16x16 f32 tiles; B operand = K^T ==> row-major
    // fragment load over K's rows (lane&15 indexes the key).
    v8f s[2];
#pragma unroll
    for (int ns = 0; ns < 2; ++ns) {
      s[ns] = v8f{};
#pragma unroll
      for (int kc = 0; kc < 2; ++kc) {
        v16h kf = load_frag_rowmajor(Kh + (size_t)(kb + ns * 16) * HDIM + kc * 32, HDIM);
        s[ns] = __builtin_amdgcn_wmma_f32_16x16x32_f16(
            false, qf[kc], false, kf, (short)0, s[ns], false, false);
      }
    }

    // Mask penalties (block identity + key nodal mask).
#pragma unroll
    for (int ns = 0; ns < 2; ++ns) {
      const int srcLane = ns * 16 + nl;
      const int   cbv = __shfl(cb, srcLane, 32);
      const float cnv = __shfl(cnod, srcLane, 32);
#pragma unroll
      for (int r = 0; r < 8; ++r) {
        float pen = ((cbv == rowblk[r]) ? 0.0f : NEG_INF_F) +
                    ((cnv > 0.5f) ? 0.0f : NEG_INF_F);
        s[ns][r] += pen;
      }
    }

    // Online softmax (each score row lives across the 16 lanes of its half).
#pragma unroll
    for (int r = 0; r < 8; ++r) {
      float tmax = fmaxf(s[0][r], s[1][r]);
#pragma unroll
      for (int off = 8; off > 0; off >>= 1)
        tmax = fmaxf(tmax, __shfl_xor(tmax, off, 16));
      const float mnew = fmaxf(mrow[r], tmax);
      const float sc = __expf(mrow[r] - mnew);
      const float p0 = __expf(s[0][r] - mnew);
      const float p1 = __expf(s[1][r] - mnew);
      float psum = p0 + p1;
#pragma unroll
      for (int off = 8; off > 0; off >>= 1)
        psum += __shfl_xor(psum, off, 16);
      lrow[r] = lrow[r] * sc + psum;
      mrow[r] = mnew;
#pragma unroll
      for (int dt = 0; dt < 4; ++dt) acc[dt][r] *= sc;
      // C-layout -> row-major LDS bounce for the A-operand of P@V.
      ldsP[(r + 8 * half) * 32 + nl]      = (_Float16)p0;
      ldsP[(r + 8 * half) * 32 + 16 + nl] = (_Float16)p1;
    }
    __syncthreads();

    v16h pf = load_frag_rowmajor(ldsP, 32);
#pragma unroll
    for (int dt = 0; dt < 4; ++dt) {
      // B operand = V tile (32 keys x 16 d); V^T rows are d, contiguous keys.
      v16h vf = load_frag_rowmajor(Vth + (size_t)(dt * 16) * SEQ + kb, SEQ);
      acc[dt] = __builtin_amdgcn_wmma_f32_16x16x32_f16(
          false, pf, false, vf, (short)0, acc[dt], false, false);
    }
    __syncthreads();
  }

  // Normalize and write attention output as (B, N, UNITS) f16.
#pragma unroll
  for (int r = 0; r < 8; ++r) {
    const float inv = (lrow[r] > 0.0f) ? (1.0f / lrow[r]) : 0.0f;
    const size_t row = (size_t)b * SEQ + rowBase + r + 8 * half;
#pragma unroll
    for (int dt = 0; dt < 4; ++dt)
      outAttn[row * UNITS_ + h * HDIM + dt * 16 + nl] = (_Float16)(acc[dt][r] * inv);
  }
}

// ---------------------------------------------------------------------------
// Launch
// ---------------------------------------------------------------------------
extern "C" void kernel_launch(void* const* d_in, const int* in_sizes, int n_in,
                              void* d_out, int out_size, void* d_ws, size_t ws_size,
                              hipStream_t stream) {
  const float* x      = (const float*)d_in[0];
  const float* blockB = (const float*)d_in[1];
  const float* nodal  = (const float*)d_in[2];
  const float* Wq     = (const float*)d_in[3];
  const float* Wk     = (const float*)d_in[4];
  const float* Wv     = (const float*)d_in[5];
  const float* Wo     = (const float*)d_in[6];
  float* out = (float*)d_out;

  // Workspace layout (all 256B aligned). Total ~48.02 MB.
  char* ws = (char*)d_ws;
  size_t off = 0;
  auto take = [&](size_t bytes) { char* p = ws + off; off = (off + bytes + 255) & ~(size_t)255; return p; };
  _Float16* xh    = (_Float16*)take((size_t)MROWS * UNITS_ * 2);
  _Float16* wqt   = (_Float16*)take((size_t)UNITS_ * UNITS_ * 2);
  _Float16* wkt   = (_Float16*)take((size_t)UNITS_ * UNITS_ * 2);
  _Float16* wvt   = (_Float16*)take((size_t)UNITS_ * UNITS_ * 2);
  _Float16* wot   = (_Float16*)take((size_t)UNITS_ * UNITS_ * 2);
  _Float16* qh    = (_Float16*)take((size_t)MROWS * UNITS_ * 2);
  _Float16* kh    = (_Float16*)take((size_t)MROWS * UNITS_ * 2);
  _Float16* vth   = (_Float16*)take((size_t)MROWS * UNITS_ * 2);
  _Float16* attnh = (_Float16*)take((size_t)MROWS * UNITS_ * 2);
  int*      ids   = (int*)take((size_t)MROWS * 4);

  // 1) fp32 -> fp16 conversion of x; weights converted + transposed (N-major).
  {
    const int nx = MROWS * UNITS_;
    f32_to_f16_kernel<<<(nx + 255) / 256, 256, 0, stream>>>(x, xh, nx);
    dim3 tgrid(UNITS_ / 32, UNITS_ / 32);
    transpose_f32_to_f16_kernel<<<tgrid, 256, 0, stream>>>(Wq, wqt, UNITS_, UNITS_);
    transpose_f32_to_f16_kernel<<<tgrid, 256, 0, stream>>>(Wk, wkt, UNITS_, UNITS_);
    transpose_f32_to_f16_kernel<<<tgrid, 256, 0, stream>>>(Wv, wvt, UNITS_, UNITS_);
    transpose_f32_to_f16_kernel<<<tgrid, 256, 0, stream>>>(Wo, wot, UNITS_, UNITS_);
  }

  // 2) one-hot -> block ids.
  block_id_kernel<<<(MROWS + 255) / 256, 256, 0, stream>>>(blockB, ids, MROWS);

  // 3) Q/K/V projections. 1/sqrt(HDIM) folded into Q. V stored transposed.
  dim3 ggrid(MROWS / 64, UNITS_ / 64);
  gemm_f16_kernel<<<ggrid, 128, 0, stream>>>(xh, wqt, nullptr, qh,
                                             MROWS, UNITS_, UNITS_, 2, 0.125f);
  gemm_f16_kernel<<<ggrid, 128, 0, stream>>>(xh, wkt, nullptr, kh,
                                             MROWS, UNITS_, UNITS_, 2, 1.0f);
  gemm_f16_kernel<<<ggrid, 128, 0, stream>>>(xh, wvt, nullptr, vth,
                                             MROWS, UNITS_, UNITS_, 3, 1.0f);

  // 4) masked flash attention.
  flash_attn_kernel<<<BATCH * NHEADS * (SEQ / 16), 32, 0, stream>>>(
      qh, kh, vth, ids, nodal, attnh);

  // 5) output projection (f32 result straight to d_out).
  gemm_f16_kernel<<<ggrid, 128, 0, stream>>>(attnh, wot, out, nullptr,
                                             MROWS, UNITS_, UNITS_, 0, 1.0f);
}
